// LCMC_37160057045707
// MI455X (gfx1250) — compile-verified, matrix operand order checked by reading, not run on hardware
//
#include <hip/hip_runtime.h>

typedef _Float16 v16h __attribute__((ext_vector_type(16)));
typedef _Float16 v8h  __attribute__((ext_vector_type(8)));
typedef _Float16 v4h  __attribute__((ext_vector_type(4)));
typedef float    v8f  __attribute__((ext_vector_type(8)));

#define CIN   256
#define COUT  256
#define HW    64
#define LTOT  4096           // 64*64 pixels
#define KTOT  2304           // CIN * 9
#define KB_N  72             // KTOT / 32
#define NPIX  32             // pixels per workgroup (half an image row)
#define LDS_STRIDE 40        // halves per LDS row: 32 + 8 pad (80B -> 16B aligned, bank-spread)
#define PF_DIST 3            // prefetch distance in K-steps for the mask stream

// ---- weights f32 -> f16 (index preserving: wf[o*2304 + k]) ----
__global__ __launch_bounds__(256) void wcvt_kernel(const float* __restrict__ w,
                                                   _Float16* __restrict__ o) {
    int i = blockIdx.x * 256 + threadIdx.x;
    o[i] = (_Float16)w[i];
}

// Staged element: k is WAVE-UNIFORM (scalar) -> all unfold math + row bases on SALU.
// Only col / clamp / select / mul are per-lane VALU.
__device__ __forceinline__ void stage_calc(int k, int y0, int colbase, int sp,
                                           const float* __restrict__ xB,
                                           const float* __restrict__ maskBase,
                                           float& mv, float& xv)
{
    int c = k / 9;  int r = k - c * 9;        // scalar (SALU magic-multiply)
    int i = r / 3;  int j = r - i * 3;        // scalar
    int yy  = y0 + i - 1;                     // scalar
    int yyc = yy < 0 ? 0 : (yy > 63 ? 63 : yy);
    bool yv = (unsigned)yy < 64u;             // scalar row-validity
    const float* xrow = xB + ((size_t)c << 12) + ((size_t)yyc << 6);  // scalar base
    const float* mrow = maskBase + ((size_t)k << 12);                 // scalar base
    int col  = colbase + j;                   // lane-varying
    int colc = col & 63;                      // maps -1->63, 64->0 : same row, valid memory
    mv = mrow[sp];                            // saddr + loop-invariant voffset
    float xl = xrow[colc];
    xv = (yv && (unsigned)col < 64u) ? xl : 0.0f;   // single cndmask, scalar part folded
}

// ---- fused unfold * mask -> f16 panel in LDS -> WMMA GEMM ----
__global__ __launch_bounds__(256) void mconv_kernel(const float* __restrict__ x,
                                                    const float* __restrict__ mask,
                                                    const _Float16* __restrict__ wf,
                                                    float* __restrict__ out)
{
    __shared__ __align__(16) _Float16 smB[2][NPIX * LDS_STRIDE];

    const int t    = threadIdx.x;
    const int lane = t & 31;
    const int l0   = blockIdx.x * NPIX;    // pixel block start
    const int b    = blockIdx.y;           // batch
    const int y0   = l0 >> 6;              // image row
    const int xx0  = l0 & 63;              // 0 or 32

    // wave index is uniform: force it into an SGPR so the k->(c,i,j) math scalarizes
    const int wv  = __builtin_amdgcn_readfirstlane(t >> 5);   // wave 0..7 == k-quad
    const int sp  = t & 31;                // staged pixel 0..31 (== lane)
    const int colbase = xx0 + sp - 1;      // per-lane column base (j added later)

    const float* maskBase = mask + (size_t)b * KTOT * LTOT + l0;
    const float* xB       = x + (size_t)b * CIN * HW * HW;

    // ---- stage K-step 0 into buffer 0 ----
    {
        v4h hv;
        #pragma unroll
        for (int q = 0; q < 4; ++q) {
            float mv, xv;
            stage_calc(wv * 4 + q, y0, colbase, sp, xB, maskBase, mv, xv);
            hv[q] = (_Float16)(xv * mv);
        }
        *(v4h*)&smB[0][sp * LDS_STRIDE + wv * 4] = hv;
    }
    __syncthreads();

    v8f acc00 = {}, acc01 = {}, acc10 = {}, acc11 = {};

    // A-fragment addressing: lane 0-15 -> M row m, K 0..7 & 16..23; lane 16-31 -> K 8..15 & 24..31
    const int mrow = wv * 32 + (lane & 15);
    const int koff = (lane >> 4) << 3;     // 0 or 8 halves
    const _Float16* wBase  = wf + (size_t)mrow * KTOT + koff;
    const _Float16* wBase2 = wBase + (size_t)16 * KTOT;

    int buf = 0;
    for (int kb = 0; kb < KB_N; ++kb) {
        // --- deep prefetch of the mask stream (global_prefetch_b8, no counters) ---
        if (kb + PF_DIST < KB_N) {
            #pragma unroll
            for (int q = 0; q < 4; ++q) {
                int kpf = (kb + PF_DIST) * 32 + wv * 4 + q;      // scalar
                __builtin_prefetch(maskBase + ((size_t)kpf << 12) + sp, 0, 1);
            }
        }

        // --- prefetch next panel's globals into registers (overlaps WMMA) ---
        float mv[4], xv[4];
        if (kb + 1 < KB_N) {
            #pragma unroll
            for (int q = 0; q < 4; ++q)
                stage_calc((kb + 1) * 32 + wv * 4 + q, y0, colbase, sp, xB, maskBase,
                           mv[q], xv[q]);
        }

        // --- A fragments (f16 weights, L2-resident), two 16B loads each ---
        const _Float16* wp  = wBase  + (size_t)kb * 32;
        const _Float16* wp2 = wBase2 + (size_t)kb * 32;
        v8h a0lo = *(const v8h*)(wp);
        v8h a0hi = *(const v8h*)(wp + 16);
        v8h a1lo = *(const v8h*)(wp2);
        v8h a1hi = *(const v8h*)(wp2 + 16);
        v16h a0 = __builtin_shufflevector(a0lo, a0hi, 0,1,2,3,4,5,6,7,8,9,10,11,12,13,14,15);
        v16h a1 = __builtin_shufflevector(a1lo, a1hi, 0,1,2,3,4,5,6,7,8,9,10,11,12,13,14,15);

        // --- B fragments from LDS (transposed panel: row = pixel, K contiguous) ---
        const _Float16* s0 = &smB[buf][(lane & 15) * LDS_STRIDE + koff];        // pixels 0-15
        const _Float16* s1 = &smB[buf][((lane & 15) + 16) * LDS_STRIDE + koff]; // pixels 16-31
        v8h b0lo = *(const v8h*)s0;
        v8h b0hi = *(const v8h*)(s0 + 16);
        v8h b1lo = *(const v8h*)s1;
        v8h b1hi = *(const v8h*)(s1 + 16);
        v16h b0 = __builtin_shufflevector(b0lo, b0hi, 0,1,2,3,4,5,6,7,8,9,10,11,12,13,14,15);
        v16h b1 = __builtin_shufflevector(b1lo, b1hi, 0,1,2,3,4,5,6,7,8,9,10,11,12,13,14,15);

        // --- 4 WMMAs: 32 Cout x 32 px per wave per K step ---
        acc00 = __builtin_amdgcn_wmma_f32_16x16x32_f16(false, a0, false, b0, (short)0, acc00, false, false);
        acc01 = __builtin_amdgcn_wmma_f32_16x16x32_f16(false, a0, false, b1, (short)0, acc01, false, false);
        acc10 = __builtin_amdgcn_wmma_f32_16x16x32_f16(false, a1, false, b0, (short)0, acc10, false, false);
        acc11 = __builtin_amdgcn_wmma_f32_16x16x32_f16(false, a1, false, b1, (short)0, acc11, false, false);

        // --- commit next panel to the other buffer ---
        if (kb + 1 < KB_N) {
            v4h hv;
            #pragma unroll
            for (int q = 0; q < 4; ++q) hv[q] = (_Float16)(xv[q] * mv[q]);
            *(v4h*)&smB[buf ^ 1][sp * LDS_STRIDE + wv * 4] = hv;
        }
        __syncthreads();
        buf ^= 1;
    }

    // ---- epilogue: C/D layout — lane 0-15: N=lane, M=vgpr; lane 16-31: M=vgpr+8 ----
    const int m0   = wv * 32;
    const int vofs = (lane >> 4) << 3;   // 0 or 8
    const int n    = lane & 15;
    float* ob = out + ((size_t)b * COUT + m0 + vofs) * LTOT + l0 + n;
    #pragma unroll
    for (int v = 0; v < 8; ++v) {
        ob[(size_t)v * LTOT]             = acc00[v];   // M tile m0,    pixels 0-15
        ob[(size_t)v * LTOT + 16]        = acc01[v];   // M tile m0,    pixels 16-31
        ob[((size_t)v + 16) * LTOT]      = acc10[v];   // M tile m0+16, pixels 0-15
        ob[((size_t)v + 16) * LTOT + 16] = acc11[v];   // M tile m0+16, pixels 16-31
    }
}

extern "C" void kernel_launch(void* const* d_in, const int* in_sizes, int n_in,
                              void* d_out, int out_size, void* d_ws, size_t ws_size,
                              hipStream_t stream) {
    (void)in_sizes; (void)n_in; (void)out_size; (void)ws_size;
    const float* x    = (const float*)d_in[0];   // [8,256,64,64]
    const float* mask = (const float*)d_in[1];   // [8,2304,4096]
    const float* w    = (const float*)d_in[2];   // [256,256,3,3] == [256,2304]
    _Float16* wf = (_Float16*)d_ws;              // 589824 halves = 1.13 MB scratch

    wcvt_kernel<<<(KTOT * COUT) / 256, 256, 0, stream>>>(w, wf);

    dim3 grid(LTOT / NPIX, 8);                   // 128 pixel-blocks x 8 batches
    mconv_kernel<<<grid, 256, 0, stream>>>(x, mask, wf, (float*)d_out);
}
